// InteractionBlockTripletsOnly_78408922956497
// MI455X (gfx1250) — matrix operand order repaired and checked by reading, not compiled, not run on hardware
//
#include <hip/hip_runtime.h>
#include <hip/hip_bf16.h>
#include <math.h>
#include <stdint.h>

// ---------------------------------------------------------------------------
// Types for CDNA5 WMMA / TDM (gfx1250, wave32)
// ---------------------------------------------------------------------------
typedef __attribute__((ext_vector_type(16))) __bf16 v16bf;
typedef __attribute__((ext_vector_type(8)))  __bf16 v8bf;
typedef __attribute__((ext_vector_type(4)))  __bf16 v4bf;
typedef __attribute__((ext_vector_type(8)))  float  v8f;
typedef __attribute__((ext_vector_type(4)))  unsigned int v4u;
typedef __attribute__((ext_vector_type(8)))  int v8i;
typedef __attribute__((ext_vector_type(4)))  int v4i;

#if defined(__has_builtin)
#  if __has_builtin(__builtin_amdgcn_tensor_load_to_lds)
#    define HAS_TDM 1
#  endif
#endif
#ifndef HAS_TDM
#  define HAS_TDM 0
#endif

#define INV_SQRT2 0.70710678118654752440f

__device__ __forceinline__ float scaled_silu(float x) {
    return (x / (1.0f + __expf(-x))) * (1.0f / 0.6f);
}

__device__ __forceinline__ v16bf cat16(v8bf lo, v8bf hi) {
    return __builtin_shufflevector(lo, hi, 0, 1, 2, 3, 4, 5, 6, 7,
                                           8, 9, 10, 11, 12, 13, 14, 15);
}

__device__ __forceinline__ v4bf cvt4(float4 f) {
    v4bf r;
    r[0] = (__bf16)f.x; r[1] = (__bf16)f.y;
    r[2] = (__bf16)f.z; r[3] = (__bf16)f.w;
    return r;
}

// ---------------------------------------------------------------------------
// Weight pre-pack: Wt[n*K+k] = bf16(W[k*N+n])
// ---------------------------------------------------------------------------
__global__ void pack_wt_kernel(const float* __restrict__ W,
                               __bf16* __restrict__ Wt, int K, int N)
{
    size_t total = (size_t)K * N;
    size_t stride = (size_t)gridDim.x * blockDim.x;
    for (size_t i = (size_t)blockIdx.x * blockDim.x + threadIdx.x;
         i < total; i += stride) {
        size_t n = i / (size_t)K;
        size_t k = i - n * (size_t)K;
        Wt[i] = (__bf16)W[k * (size_t)N + n];
    }
}

// ---------------------------------------------------------------------------
// TDM descriptor issue: DMA a 128-row x 32-elem bf16 tile from Wt (row
// pitch K elements) into LDS with 16B padding after each 64B row
// (matching LDS_STRIDE = 40 elements).
// 6-arg builtin form: (v4u g0, v8i g1, v4i g2, v4i g3, v8i g4, i32 cpol).
// ---------------------------------------------------------------------------
#if HAS_TDM
__device__ __forceinline__ void tdm_load_b_tile(const __bf16* gsrc,
                                                unsigned ldsAddr, int K)
{
    unsigned long long ga = (unsigned long long)(uintptr_t)gsrc;
    v4u g0;
    g0[0] = 1u;                                   // count=1, user mode
    g0[1] = ldsAddr;                              // lds byte address
    g0[2] = (unsigned)ga;                         // global_addr[31:0]
    g0[3] = (unsigned)(ga >> 32) | (2u << 30);    // global_addr[56:32] | type=2
    v8i g1;
    g1[0] = (int)((1u << 16)    // data_size = 2 bytes
                | (1u << 20)    // pad_enable
                | (3u << 22)    // pad_interval: 16 dwords (64B of row data)
                | (3u << 25));  // pad_amount:   4 dwords (16B pad)
    unsigned K_ = (unsigned)K;
    g1[1] = (int)((K_ & 0xFFFFu) << 16);                       // tensor_dim0 lo16
    g1[2] = (int)(((K_ >> 16) & 0xFFFFu) | (128u << 16));      // td0 hi16 | tensor_dim1 lo16 (=128)
    g1[3] = (int)(32u << 16);                                  // td1 hi16 (0) | tile_dim0 = 32
    g1[4] = 128;                                               // tile_dim1 = 128 (tile_dim2 = 0)
    g1[5] = (int)K_;                                           // tensor_dim0_stride lo32
    g1[6] = 0;                                                 // stride hi16 | td1_stride lo16
    g1[7] = 0;
    v4i z4 = {0, 0, 0, 0};
    v8i z8 = {0, 0, 0, 0, 0, 0, 0, 0};
    __builtin_amdgcn_tensor_load_to_lds(g0, g1, z4, z4, z8, 0);
}
#endif

// ---------------------------------------------------------------------------
// WMMA GEMM:  Y = epilogue(X @ W)
//   X : (M,K) f32 row-major        (converted to bf16 on LDS fill)
//   Wt: (N,K) bf16 row-major       (pre-packed, pre-transposed)
//   MODE 0: y = act(s)
//   MODE 1: y = (R[idx] + act(s)) * inv_sqrt2
//   MODE 2: y = act(s) * R[idx]
// K multiple of 32 (all call sites). Ping-pong LDS, one barrier per K-step.
// B tiles arrive via TDM (tensor_load_to_lds) when available.
// ---------------------------------------------------------------------------
#define GEMM_BM 128
#define GEMM_BN 128
#define GEMM_BK 32
#define LDS_STRIDE 40   // 80B row pitch (64B data + 16B pad)

template <int MODE>
__global__ __launch_bounds__(256) void gemm_wmma_kernel(
    const float* __restrict__ Aptr, const __bf16* __restrict__ Wt,
    float* __restrict__ Yptr, const float* __restrict__ Rptr,
    int M, int N, int K)
{
    __shared__ __align__(16) __bf16 As[2][GEMM_BM][LDS_STRIDE];
    __shared__ __align__(16) __bf16 Bs[2][GEMM_BN][LDS_STRIDE]; // [n][k]

    const int tid  = threadIdx.x;
    const int lane = tid & 31;
    const int wave = tid >> 5;
    const int wm   = wave & 3;   // 4 waves along M
    const int wn   = wave >> 2;  // 2 waves along N

    const int blockM = blockIdx.y * GEMM_BM;
    const int blockN = blockIdx.x * GEMM_BN;

    const int rowInWave = lane & 15;
    const int khalf     = lane >> 4;

    // ---- A staging descriptors ----
    int aR_[4], aC_[4];
    const float* aBase[4];
#pragma unroll
    for (int i = 0; i < 4; ++i) {
        int idx4 = tid + i * 256;
        int r = idx4 >> 3;
        int c = (idx4 & 7) * 4;
        aR_[i] = r; aC_[i] = c;
        int gr = blockM + r; if (gr > M - 1) gr = M - 1;   // clamp: safe
        aBase[i] = Aptr + (size_t)gr * K + c;
    }
#if !HAS_TDM
    const int bn = tid >> 1;
    const int bh = (tid & 1) * 16;
    int gn = blockN + bn; if (gn > N - 1) gn = N - 1;
    const __bf16* bBase = Wt + (size_t)gn * K + bh;
    float4 bReg[2];
#endif
    const __bf16* bTile = Wt + (size_t)blockN * K;   // tile row 0 (TDM src)

    v8f acc[2][4];
    const v8f vzero = {0.f, 0.f, 0.f, 0.f, 0.f, 0.f, 0.f, 0.f};
#pragma unroll
    for (int tm = 0; tm < 2; ++tm)
#pragma unroll
        for (int tn = 0; tn < 4; ++tn) acc[tm][tn] = vzero;

    float4 aReg[4];

    // ---- prologue: stage tile 0 into buffer 0 ----
#if HAS_TDM
    if (tid < 32) {
        tdm_load_b_tile(bTile, (unsigned)(uintptr_t)&Bs[0][0][0], K);
    }
#else
    bReg[0] = *reinterpret_cast<const float4*>(bBase);
    bReg[1] = *reinterpret_cast<const float4*>(bBase + 8);
    *reinterpret_cast<float4*>(&Bs[0][bn][bh])     = bReg[0];
    *reinterpret_cast<float4*>(&Bs[0][bn][bh + 8]) = bReg[1];
#endif
#pragma unroll
    for (int i = 0; i < 4; ++i)
        aReg[i] = *reinterpret_cast<const float4*>(aBase[i]);
#pragma unroll
    for (int i = 0; i < 4; ++i)
        *reinterpret_cast<v4bf*>(&As[0][aR_[i]][aC_[i]]) = cvt4(aReg[i]);
#if HAS_TDM
    if (tid < 32) __builtin_amdgcn_s_wait_tensorcnt(0);
#endif
    __syncthreads();

    int cur = 0;
    for (int k0 = 0; k0 < K; k0 += GEMM_BK) {
        const int kn  = k0 + GEMM_BK;
        const bool more = (kn < K);
        const int nxt = cur ^ 1;

        if (more) {
            // issue next B tile DMA + next A global loads (latency hidden
            // behind this iteration's WMMAs)
#if HAS_TDM
            if (tid < 32)
                tdm_load_b_tile(bTile + kn,
                                (unsigned)(uintptr_t)&Bs[nxt][0][0], K);
#else
            bReg[0] = *reinterpret_cast<const float4*>(bBase + kn);
            bReg[1] = *reinterpret_cast<const float4*>(bBase + kn + 8);
#endif
#pragma unroll
            for (int i = 0; i < 4; ++i)
                aReg[i] = *reinterpret_cast<const float4*>(aBase[i] + kn);
        }

        // ---- fragments + WMMA on current tiles ----
        v16bf afrag[2];
#pragma unroll
        for (int tm = 0; tm < 2; ++tm) {
            int row = wm * 32 + tm * 16 + rowInWave;
            v8bf lo = *reinterpret_cast<const v8bf*>(&As[cur][row][khalf * 8]);
            v8bf hi = *reinterpret_cast<const v8bf*>(&As[cur][row][16 + khalf * 8]);
            afrag[tm] = cat16(lo, hi);
        }
        v16bf bfrag[4];
#pragma unroll
        for (int tn = 0; tn < 4; ++tn) {
            int col = wn * 64 + tn * 16 + rowInWave;
            int kb  = khalf * 8;
            v8bf lo = *reinterpret_cast<const v8bf*>(&Bs[cur][col][kb]);
            v8bf hi = *reinterpret_cast<const v8bf*>(&Bs[cur][col][kb + 16]);
            bfrag[tn] = cat16(lo, hi);
        }
#pragma unroll
        for (int tm = 0; tm < 2; ++tm)
#pragma unroll
            for (int tn = 0; tn < 4; ++tn)
                acc[tm][tn] = __builtin_amdgcn_wmma_f32_16x16x32_bf16(
                    false, afrag[tm], false, bfrag[tn],
                    (short)0, acc[tm][tn], false, false);

        if (more) {
            // commit staged data into the 'next' buffers (nobody reads them
            // until after the barrier below)
#pragma unroll
            for (int i = 0; i < 4; ++i)
                *reinterpret_cast<v4bf*>(&As[nxt][aR_[i]][aC_[i]]) = cvt4(aReg[i]);
#if HAS_TDM
            if (tid < 32) __builtin_amdgcn_s_wait_tensorcnt(0);
#else
            *reinterpret_cast<float4*>(&Bs[nxt][bn][bh])     = bReg[0];
            *reinterpret_cast<float4*>(&Bs[nxt][bn][bh + 8]) = bReg[1];
#endif
        }
        __syncthreads();
        cur = nxt;
    }

    // ---- epilogue ----
#pragma unroll
    for (int tm = 0; tm < 2; ++tm) {
#pragma unroll
        for (int tn = 0; tn < 4; ++tn) {
#pragma unroll
            for (int r = 0; r < 8; ++r) {
                int row = blockM + wm * 32 + tm * 16 + r + 8 * khalf;
                int col = blockN + wn * 64 + tn * 16 + rowInWave;
                if (row < M && col < N) {
                    size_t idx = (size_t)row * N + col;
                    float s = acc[tm][tn][r];
                    float y;
                    if (MODE == 0) {
                        y = scaled_silu(s);
                    } else if (MODE == 1) {
                        y = (Rptr[idx] + scaled_silu(s)) * INV_SQRT2;
                    } else {
                        y = scaled_silu(s) * Rptr[idx];
                    }
                    Yptr[idx] = y;
                }
            }
        }
    }
}

// ---------------------------------------------------------------------------
// Small-K (K=16) projection: out = rbf @ W    (nE,16)@(16,512)
// ---------------------------------------------------------------------------
__global__ __launch_bounds__(256) void rbf_proj_kernel(
    const float* __restrict__ rbf, const float* __restrict__ W,
    float* __restrict__ out, int nE)
{
    __shared__ float Ws[16][512];
    const int tid = threadIdx.x;
    for (int idx = tid; idx < 16 * 512; idx += 256)
        Ws[idx >> 9][idx & 511] = W[idx];
    __syncthreads();

    int e0 = blockIdx.x * 16;
    int e1 = (e0 + 16 < nE) ? e0 + 16 : nE;
    for (int e = e0; e < e1; ++e) {
        float rb[16];
#pragma unroll
        for (int r = 0; r < 16; ++r) rb[r] = rbf[(size_t)e * 16 + r];
        for (int c = tid; c < 512; c += 256) {
            float acc = 0.f;
#pragma unroll
            for (int r = 0; r < 16; ++r) acc += rb[r] * Ws[r][c];
            out[(size_t)e * 512 + c] = acc;
        }
    }
}

// ---------------------------------------------------------------------------
// Fused triplet bilinear. One 64-thread block per edge.
// m2[e,k,:] = xba_down[id3_expand_ba[e*10+k], :]  (reduce/Kidx3 structure)
// ---------------------------------------------------------------------------
__global__ __launch_bounds__(64) void triplet_kernel(
    const float* __restrict__ xba_down,
    const float* __restrict__ cbf_sph,
    const float* __restrict__ cbf_W1,
    const float* __restrict__ W_bil,
    const int*   __restrict__ id3_expand_ba,
    float* __restrict__ xout)
{
    const int e = blockIdx.x;
    const int t = threadIdx.x;

    __shared__ float m2[10][64];
    __shared__ float sph[10][7];
    __shared__ float w1s[16][7];
    __shared__ float sumk[7][64];
    __shared__ float rw[16][64];
    __shared__ float red[16][64];

#pragma unroll
    for (int k = 0; k < 10; ++k) {
        int src = id3_expand_ba[(size_t)e * 10 + k];
        m2[k][t] = xba_down[(size_t)src * 64 + t];
    }
    for (int idx = t; idx < 70; idx += 64)
        sph[idx / 7][idx % 7] = cbf_sph[(size_t)e * 70 + idx];
    for (int idx = t; idx < 112; idx += 64)
        w1s[idx / 7][idx % 7] = cbf_W1[(size_t)e * 112 + idx];
    __syncthreads();

#pragma unroll
    for (int s = 0; s < 7; ++s) {
        float acc = 0.f;
#pragma unroll
        for (int k = 0; k < 10; ++k) acc += sph[k][s] * m2[k][t];
        sumk[s][t] = acc;
    }
    __syncthreads();

#pragma unroll
    for (int j = 0; j < 16; ++j) {
        float acc = 0.f;
#pragma unroll
        for (int s = 0; s < 7; ++s) acc += w1s[j][s] * sumk[s][t];
        rw[j][t] = acc;
    }
    __syncthreads();

    float p[16];
#pragma unroll
    for (int o = 0; o < 16; ++o) p[o] = 0.f;
#pragma unroll
    for (int j = 0; j < 16; ++j) {
        float r = rw[j][t];
        const float* wb = &W_bil[((size_t)t * 16 + j) * 16];
#pragma unroll
        for (int o = 0; o < 16; ++o) p[o] += r * wb[o];
    }
#pragma unroll
    for (int o = 0; o < 16; ++o) red[o][t] = p[o];
    __syncthreads();

    if (t < 16) {
        float acc = 0.f;
#pragma unroll
        for (int i2 = 0; i2 < 64; ++i2) acc += red[t][i2];
        xout[(size_t)e * 16 + t] = acc;
    }
}

// ---------------------------------------------------------------------------
// x3[e] = (act(x[e]@W_up_ca) + act(x[id_swap[e]]@W_up_ac)) * inv_sqrt2
// ---------------------------------------------------------------------------
__global__ __launch_bounds__(256) void upproj_kernel(
    const float* __restrict__ xt, const float* __restrict__ Wca,
    const float* __restrict__ Wac, const int* __restrict__ id_swap,
    float* __restrict__ x3, int nE)
{
    __shared__ float Ws[2][16][512];   // 64 KB
    const int tid = threadIdx.x;
    for (int idx = tid; idx < 16 * 512; idx += 256) {
        Ws[0][idx >> 9][idx & 511] = Wca[idx];
        Ws[1][idx >> 9][idx & 511] = Wac[idx];
    }
    __syncthreads();

    int e0 = blockIdx.x * 8;
    int e1 = (e0 + 8 < nE) ? e0 + 8 : nE;
    for (int e = e0; e < e1; ++e) {
        int sw = id_swap[e];
        float xa[16], xb[16];
#pragma unroll
        for (int r = 0; r < 16; ++r) {
            xa[r] = xt[(size_t)e * 16 + r];
            xb[r] = xt[(size_t)sw * 16 + r];
        }
        for (int c = tid; c < 512; c += 256) {
            float s0 = 0.f, s1 = 0.f;
#pragma unroll
            for (int r = 0; r < 16; ++r) {
                s0 += xa[r] * Ws[0][r][c];
                s1 += xb[r] * Ws[1][r][c];
            }
            x3[(size_t)e * 512 + c] =
                (scaled_silu(s0) + scaled_silu(s1)) * INV_SQRT2;
        }
    }
}

__global__ void combine_kernel(const float* __restrict__ a,
                               const float* __restrict__ b,
                               float* __restrict__ out, size_t n)
{
    size_t i = (size_t)blockIdx.x * blockDim.x + threadIdx.x;
    size_t stride = (size_t)gridDim.x * blockDim.x;
    for (; i < n; i += stride) out[i] = (a[i] + b[i]) * INV_SQRT2;
}

__global__ void zero_kernel(float* __restrict__ p, size_t n)
{
    size_t i = (size_t)blockIdx.x * blockDim.x + threadIdx.x;
    size_t stride = (size_t)gridDim.x * blockDim.x;
    for (; i < n; i += stride) p[i] = 0.f;
}

// ---------------------------------------------------------------------------
// x2[id_a[e]] += m_new[e] * (rbf_h[e] @ W_rbf_h)
// ---------------------------------------------------------------------------
__global__ __launch_bounds__(256) void atom_scatter_kernel(
    const float* __restrict__ m_new, const float* __restrict__ rbf_h,
    const float* __restrict__ W_rbf_h, const int* __restrict__ id_a,
    float* __restrict__ x2, int nE)
{
    __shared__ float Ws[16][512];
    const int tid = threadIdx.x;
    for (int idx = tid; idx < 16 * 512; idx += 256)
        Ws[idx >> 9][idx & 511] = W_rbf_h[idx];
    __syncthreads();

    int e0 = blockIdx.x * 16;
    int e1 = (e0 + 16 < nE) ? e0 + 16 : nE;
    for (int e = e0; e < e1; ++e) {
        int a = id_a[e];
        float rb[16];
#pragma unroll
        for (int r = 0; r < 16; ++r) rb[r] = rbf_h[(size_t)e * 16 + r];
        for (int c = tid; c < 512; c += 256) {
            float rw = 0.f;
#pragma unroll
            for (int r = 0; r < 16; ++r) rw += rb[r] * Ws[r][c];
            atomicAdd(&x2[(size_t)a * 512 + c],
                      m_new[(size_t)e * 512 + c] * rw);
        }
    }
}

__global__ __launch_bounds__(256) void concat_kernel(
    const float* __restrict__ h_new, const float* __restrict__ m_new,
    const int* __restrict__ id_c, const int* __restrict__ id_a,
    float* __restrict__ out, int nE)
{
    int e = blockIdx.x;
    if (e >= nE) return;
    const float* hc = h_new + (size_t)id_c[e] * 128;
    const float* ha = h_new + (size_t)id_a[e] * 128;
    const float* mm = m_new + (size_t)e * 512;
    float* o = out + (size_t)e * 768;
    for (int c = threadIdx.x; c < 768; c += 256) {
        float v;
        if (c < 128)       v = hc[c];
        else if (c < 256)  v = ha[c - 128];
        else               v = mm[c - 256];
        o[c] = v;
    }
}

// ---------------------------------------------------------------------------
// Host-side orchestration
// ---------------------------------------------------------------------------
extern "C" void kernel_launch(void* const* d_in, const int* in_sizes, int n_in,
                              void* d_out, int out_size, void* d_ws, size_t ws_size,
                              hipStream_t stream) {
    (void)out_size; (void)ws_size;
    if (n_in < 26) return;

    const float* h          = (const float*)d_in[0];
    const float* m          = (const float*)d_in[1];
    const float* rbf3       = (const float*)d_in[2];
    const float* cbf_W1     = (const float*)d_in[3];
    const float* cbf_sph    = (const float*)d_in[4];
    const float* rbf_h      = (const float*)d_in[5];
    const float* W_dense_ca = (const float*)d_in[6];
    const float* W_ba       = (const float*)d_in[7];
    const float* W_rbf3     = (const float*)d_in[8];
    const float* W_down     = (const float*)d_in[9];
    const float* W_bil      = (const float*)d_in[10];
    const float* W_up_ca    = (const float*)d_in[11];
    const float* W_up_ac    = (const float*)d_in[12];
    const float* W_res_b    = (const float*)d_in[13];
    const float* W_res_a    = (const float*)d_in[14];
    const float* W_rbf_h    = (const float*)d_in[15];
    const float* W_atom_d   = (const float*)d_in[16];
    const float* W_atom_r   = (const float*)d_in[17];
    const float* W_concat   = (const float*)d_in[18];
    const float* W_res_m    = (const float*)d_in[19];
    const int*   id_swap    = (const int*)d_in[21];
    const int*   id3_exp    = (const int*)d_in[22];
    const int*   id_c       = (const int*)d_in[24];
    const int*   id_a       = (const int*)d_in[25];

    const int nE = in_sizes[1] / 512;
    const int nA = in_sizes[0] / 128;

    float* ws = (float*)d_ws;
    const size_t EB = (size_t)nE * 512;
    float* B0 = ws;
    float* B1 = ws + EB;
    float* B2 = ws + 2 * EB;
    float* B3 = ws + 3 * EB;   // m_new (persistent)
    float* S0 = ws + 4 * EB;                       // xba_down (nE,64)
    float* S1 = S0 + (size_t)nE * 64;              // x_trip   (nE,16)
    float* S2 = S1 + (size_t)nE * 16;              // x2       (nA,512)
    float* S3 = S2 + (size_t)nA * 512;
    float* S4 = S3 + (size_t)nA * 128;
    float* S5 = S4 + (size_t)nA * 128;

    // bf16 transposed-weight pool
    __bf16* wp = (__bf16*)(S5 + (size_t)nA * 128);
    const size_t EE = (size_t)512 * 512;
    __bf16* Wt_ba    = wp;  wp += EE;
    __bf16* Wt_dense = wp;  wp += EE;
    __bf16* Wt_down  = wp;  wp += (size_t)64 * 512;
    __bf16* Wt_rb0   = wp;  wp += EE;
    __bf16* Wt_rb1   = wp;  wp += EE;
    __bf16* Wt_ra[4];
    for (int i = 0; i < 4; ++i) { Wt_ra[i] = wp; wp += EE; }
    __bf16* Wt_ad    = wp;  wp += (size_t)128 * 512;
    __bf16* Wt_ar[6];
    for (int i = 0; i < 6; ++i) { Wt_ar[i] = wp; wp += (size_t)128 * 128; }
    __bf16* Wt_cat   = wp;  wp += (size_t)512 * 768;
    __bf16* Wt_rm0   = wp;  wp += EE;
    __bf16* Wt_rm1   = wp;  wp += EE;

    float* h_new = (float*)d_out;
    float* m_out = (float*)d_out + (size_t)nA * 128;

    const dim3 blk(256);
    auto ggrid = [](int M, int N) {
        return dim3((unsigned)((N + GEMM_BN - 1) / GEMM_BN),
                    (unsigned)((M + GEMM_BM - 1) / GEMM_BM));
    };

    // ---- weight pre-pack (transpose + bf16) ----
    pack_wt_kernel<<<512, blk, 0, stream>>>(W_ba, Wt_ba, 512, 512);
    pack_wt_kernel<<<512, blk, 0, stream>>>(W_dense_ca, Wt_dense, 512, 512);
    pack_wt_kernel<<<128, blk, 0, stream>>>(W_down, Wt_down, 512, 64);
    pack_wt_kernel<<<512, blk, 0, stream>>>(W_res_b, Wt_rb0, 512, 512);
    pack_wt_kernel<<<512, blk, 0, stream>>>(W_res_b + EE, Wt_rb1, 512, 512);
    for (int i = 0; i < 4; ++i)
        pack_wt_kernel<<<512, blk, 0, stream>>>(W_res_a + (size_t)i * EE, Wt_ra[i], 512, 512);
    pack_wt_kernel<<<256, blk, 0, stream>>>(W_atom_d, Wt_ad, 512, 128);
    for (int i = 0; i < 6; ++i)
        pack_wt_kernel<<<64, blk, 0, stream>>>(W_atom_r + (size_t)i * 128 * 128, Wt_ar[i], 128, 128);
    pack_wt_kernel<<<512, blk, 0, stream>>>(W_concat, Wt_cat, 768, 512);
    pack_wt_kernel<<<512, blk, 0, stream>>>(W_res_m, Wt_rm0, 512, 512);
    pack_wt_kernel<<<512, blk, 0, stream>>>(W_res_m + EE, Wt_rm1, 512, 512);

    // 1) rbfW3 = rbf3 @ W_rbf3 -> B0
    rbf_proj_kernel<<<(nE + 15) / 16, blk, 0, stream>>>(rbf3, W_rbf3, B0, nE);
    // 2) x_ba = act(m @ W_ba) * rbfW3 -> B1
    gemm_wmma_kernel<2><<<ggrid(nE, 512), blk, 0, stream>>>(m, Wt_ba, B1, B0, nE, 512, 512);
    // 3) x_ba_down = act(x_ba @ W_down) -> S0
    gemm_wmma_kernel<0><<<ggrid(nE, 64), blk, 0, stream>>>(B1, Wt_down, S0, nullptr, nE, 64, 512);
    // 4) triplet bilinear -> x_trip (S1)
    triplet_kernel<<<nE, dim3(64), 0, stream>>>(S0, cbf_sph, cbf_W1, W_bil, id3_exp, S1);
    // 5) x3 -> B0
    upproj_kernel<<<(nE + 7) / 8, blk, 0, stream>>>(S1, W_up_ca, W_up_ac, id_swap, B0, nE);
    // 6) x = (x3 + act(m @ W_dense_ca)) * inv -> B1
    gemm_wmma_kernel<1><<<ggrid(nE, 512), blk, 0, stream>>>(m, Wt_dense, B1, B0, nE, 512, 512);
    // 7) res_before: B1 -> B0
    gemm_wmma_kernel<0><<<ggrid(nE, 512), blk, 0, stream>>>(B1, Wt_rb0, B2, nullptr, nE, 512, 512);
    gemm_wmma_kernel<1><<<ggrid(nE, 512), blk, 0, stream>>>(B2, Wt_rb1, B0, B1, nE, 512, 512);
    // 8) m_new = (m + x) * inv -> B3
    combine_kernel<<<2048, blk, 0, stream>>>(m, B0, B3, EB);
    // 9) res_after[0]: B3 -> B1
    gemm_wmma_kernel<0><<<ggrid(nE, 512), blk, 0, stream>>>(B3, Wt_ra[0], B0, nullptr, nE, 512, 512);
    gemm_wmma_kernel<1><<<ggrid(nE, 512), blk, 0, stream>>>(B0, Wt_ra[1], B1, B3, nE, 512, 512);
    // 10) res_after[1]: B1 -> B3 (final m_new)
    gemm_wmma_kernel<0><<<ggrid(nE, 512), blk, 0, stream>>>(B1, Wt_ra[2], B0, nullptr, nE, 512, 512);
    gemm_wmma_kernel<1><<<ggrid(nE, 512), blk, 0, stream>>>(B0, Wt_ra[3], B3, B1, nE, 512, 512);
    // 11) x2 = segment_sum(m_new * (rbf_h@W_rbf_h), id_a) -> S2
    zero_kernel<<<256, blk, 0, stream>>>(S2, (size_t)nA * 512);
    atom_scatter_kernel<<<(nE + 15) / 16, blk, 0, stream>>>(B3, rbf_h, W_rbf_h, id_a, S2, nE);
    // 12) xa = act(x2 @ W_atom_dense) -> S3
    gemm_wmma_kernel<0><<<ggrid(nA, 128), blk, 0, stream>>>(S2, Wt_ad, S3, nullptr, nA, 128, 512);
    // 13) atom residual blocks: S3 -> S5
    gemm_wmma_kernel<0><<<ggrid(nA, 128), blk, 0, stream>>>(S3, Wt_ar[0], S4, nullptr, nA, 128, 128);
    gemm_wmma_kernel<1><<<ggrid(nA, 128), blk, 0, stream>>>(S4, Wt_ar[1], S5, S3, nA, 128, 128);
    gemm_wmma_kernel<0><<<ggrid(nA, 128), blk, 0, stream>>>(S5, Wt_ar[2], S4, nullptr, nA, 128, 128);
    gemm_wmma_kernel<1><<<ggrid(nA, 128), blk, 0, stream>>>(S4, Wt_ar[3], S3, S5, nA, 128, 128);
    gemm_wmma_kernel<0><<<ggrid(nA, 128), blk, 0, stream>>>(S3, Wt_ar[4], S4, nullptr, nA, 128, 128);
    gemm_wmma_kernel<1><<<ggrid(nA, 128), blk, 0, stream>>>(S4, Wt_ar[5], S5, S3, nA, 128, 128);
    // 14) h_new = (h + xa) * inv -> d_out
    combine_kernel<<<256, blk, 0, stream>>>(h, S5, h_new, (size_t)nA * 128);
    // 15) concat -> B0
    concat_kernel<<<nE, blk, 0, stream>>>(h_new, B3, id_c, id_a, B0, nE);
    // 16) mc = act(concat @ W_concat) -> B2  (K=768)
    gemm_wmma_kernel<0><<<ggrid(nE, 512), blk, 0, stream>>>(B0, Wt_cat, B2, nullptr, nE, 512, 768);
    // 17) res_m: B2 -> B1
    gemm_wmma_kernel<0><<<ggrid(nE, 512), blk, 0, stream>>>(B2, Wt_rm0, B0, nullptr, nE, 512, 512);
    gemm_wmma_kernel<1><<<ggrid(nE, 512), blk, 0, stream>>>(B0, Wt_rm1, B1, B2, nE, 512, 512);
    // 18) m_out = (m_new + mc) * inv
    combine_kernel<<<2048, blk, 0, stream>>>(B3, B1, m_out, EB);
}
